// CausalWanSelfAttention_21320217657767
// MI455X (gfx1250) — matrix operand order repaired and checked by reading, not compiled, not run on hardware
//
#include <hip/hip_runtime.h>
#include <hip/hip_bf16.h>

// ---------------------------------------------------------------------------
// CausalWanSelfAttention for MI455X (gfx1250, wave32, WMMA bf16 16x16x32)
// ---------------------------------------------------------------------------

typedef __bf16 bf16;
typedef __attribute__((ext_vector_type(16))) __bf16 v16bf;
typedef __attribute__((ext_vector_type(8)))  __bf16 v8bf;
typedef __attribute__((ext_vector_type(8)))  float  v8f;
typedef int v4i_ __attribute__((vector_size(16)));

union Frag { v16bf v; v8bf h[2]; };

#define S_TOK   1560
#define SP      1568            // S padded to multiple of 16
#define DIM_    1536
#define NH_     12
#define HD_     128
#define LKV     9360            // real keys
#define LP      9376            // padded to multiple of 32
#define NEWOFF  7800            // where new K/V land in the gathered cache
#define FRAME_  1560

// ---- CDNA5 async global->LDS copy (16B), with portable fallback -----------
__device__ __forceinline__ void async_copy16(bf16* ldst, const bf16* gsrc) {
#if __has_builtin(__builtin_amdgcn_global_load_async_to_lds_b128)
  __builtin_amdgcn_global_load_async_to_lds_b128(
      (__attribute__((address_space(1))) v4i_*)gsrc,
      (__attribute__((address_space(3))) v4i_*)ldst, 0, 0);
#else
  *(v8bf*)ldst = *(const v8bf*)gsrc;
#endif
}

__device__ __forceinline__ void wait_async() {
#if __has_builtin(__builtin_amdgcn_s_wait_asynccnt)
  __builtin_amdgcn_s_wait_asynccnt(0);
#else
  asm volatile("s_wait_asynccnt 0" ::: "memory");
#endif
  asm volatile("" ::: "memory");
}

// ---------------------------------------------------------------------------
// small conversion / layout kernels
// ---------------------------------------------------------------------------
__global__ void cvt_f32_to_bf16(const float* __restrict__ src, bf16* __restrict__ dst, int n) {
  int i = blockIdx.x * 256 + threadIdx.x;
  if (i < n) dst[i] = (bf16)src[i];
}

__global__ void cvt_x_pad(const float* __restrict__ x, bf16* __restrict__ xb) {
  int i = blockIdx.x * 256 + threadIdx.x;
  if (i >= SP * DIM_) return;
  int r = i / DIM_;
  xb[i] = (r < S_TOK) ? (bf16)x[i] : (bf16)0.0f;
}

__global__ void concat_bias(const float* __restrict__ bq, const float* __restrict__ bk,
                            const float* __restrict__ bv, float* __restrict__ b) {
  int i = blockIdx.x * 256 + threadIdx.x;
  if (i < DIM_)            b[i] = bq[i];
  else if (i < 2 * DIM_)   b[i] = bk[i - DIM_];
  else if (i < 3 * DIM_)   b[i] = bv[i - 2 * DIM_];
}

// Gather sink + rolled window of the f32 KV cache into bf16 buffers.
// K layout: [h][key][dim]   (row-major per head; tiles are contiguous)
// V layout: [h][dim][key]   (transposed so PV B-fragments are contiguous)
__global__ void gather_cache(const float* __restrict__ ck, const float* __restrict__ cv,
                             bf16* __restrict__ Kc, bf16* __restrict__ VcT) {
  long i = (long)blockIdx.x * 256 + threadIdx.x;
  const long TOT = (long)NH_ * NEWOFF * HD_;
  if (i >= TOT) return;
  int d = (int)(i % HD_);
  long r = i / HD_;
  int j = (int)(r % NEWOFF);
  int h = (int)(r / NEWOFF);
  int src = (j < 1560) ? j : j + 1560;  // [0:1560]=sink, then cache[3120:9360]
  size_t si = ((size_t)src * NH_ + h) * HD_ + d;
  Kc[(size_t)h * LP * HD_ + (size_t)j * HD_ + d]  = (bf16)ck[si];
  VcT[(size_t)h * HD_ * LP + (size_t)d * LP + j]  = (bf16)cv[si];
}

// Zero key padding [9360,9376) and Q padding rows [1560,1568).
__global__ void pad_kernel(bf16* __restrict__ Kc, bf16* __restrict__ VcT, bf16* __restrict__ Qb) {
  int i = blockIdx.x * 256 + threadIdx.x;
  if (i < NH_ * 16 * HD_) {
    int d = i % HD_; int r = i / HD_;
    int j = LKV + (r % 16); int h = r / 16;
    Kc[(size_t)h * LP * HD_ + (size_t)j * HD_ + d] = (bf16)0.0f;
    VcT[(size_t)h * HD_ * LP + (size_t)d * LP + j] = (bf16)0.0f;
  }
  if (i < NH_ * (SP - S_TOK) * HD_) {
    int d = i % HD_; int r = i / HD_;
    int m = S_TOK + (r % (SP - S_TOK)); int h = r / (SP - S_TOK);
    Qb[(size_t)h * SP * HD_ + (size_t)m * HD_ + d] = (bf16)0.0f;
  }
}

// ---------------------------------------------------------------------------
// GEMM: C[M][N] = A[M][K](bf16) @ W[N][K]^T(bf16) + bias[N], f32 accumulate.
// Block = 4 waves; each wave computes one 16(M) x 64(N) tile.
// Software-pipelined: fragments for k-step i+1 are loaded while the 4 WMMAs of
// k-step i execute (two explicit register sets, no dynamic reg indexing).
// K must be a multiple of 64.
// ---------------------------------------------------------------------------
__device__ __forceinline__ void gemm_load(const bf16* arow, const bf16* wbase, int K,
                                          int k0, Frag& ua, Frag* ub) {
  ua.h[0] = *(const v8bf*)(arow + k0);
  ua.h[1] = *(const v8bf*)(arow + k0 + 16);
#pragma unroll
  for (int t = 0; t < 4; t++) {
    const bf16* wrow = wbase + (size_t)(t * 16) * K + k0;
    ub[t].h[0] = *(const v8bf*)(wrow);
    ub[t].h[1] = *(const v8bf*)(wrow + 16);
  }
}

__global__ __launch_bounds__(128)
void gemm_bf16_wmma(const bf16* __restrict__ A, const bf16* __restrict__ W,
                    const float* __restrict__ bias, float* __restrict__ C,
                    int K, int N, int Mlimit) {
  const int lane = threadIdx.x & 31;
  const int wv   = threadIdx.x >> 5;
  const int m0   = blockIdx.y * 16;
  const int n0   = blockIdx.x * 256 + wv * 64;
  const int lr   = lane & 15, lh = lane >> 4;

  v8f acc[4];
#pragma unroll
  for (int t = 0; t < 4; t++) acc[t] = {};

  const bf16* arow  = A + (size_t)(m0 + lr) * K + lh * 8;
  const bf16* wbase = W + (size_t)(n0 + lr) * K + lh * 8;

  Frag uaA, uaB; Frag ubA[4], ubB[4];
  gemm_load(arow, wbase, K, 0, uaA, ubA);

  for (int k0 = 0; k0 + 64 <= K; k0 += 64) {
    gemm_load(arow, wbase, K, k0 + 32, uaB, ubB);
#pragma unroll
    for (int t = 0; t < 4; t++)
      acc[t] = __builtin_amdgcn_wmma_f32_16x16x32_bf16(
          false, uaA.v, false, ubA[t].v, (short)0, acc[t], false, false);
    if (k0 + 64 < K) gemm_load(arow, wbase, K, k0 + 64, uaA, ubA);
#pragma unroll
    for (int t = 0; t < 4; t++)
      acc[t] = __builtin_amdgcn_wmma_f32_16x16x32_bf16(
          false, uaB.v, false, ubB[t].v, (short)0, acc[t], false, false);
  }

#pragma unroll
  for (int t = 0; t < 4; t++) {
#pragma unroll
    for (int r = 0; r < 8; r++) {
      int row = m0 + r + 8 * lh;
      int col = n0 + t * 16 + lr;
      if (row < Mlimit) C[(size_t)row * N + col] = acc[t][r] + bias[col];
    }
  }
}

// ---------------------------------------------------------------------------
// RMSNorm(q,k) + RoPE + scatter of new K/V into gathered cache buffers.
// ---------------------------------------------------------------------------
__global__ __launch_bounds__(256)
void norm_rope_kernel(const float* __restrict__ qkv, const float* __restrict__ nqw,
                      const float* __restrict__ nkw, const float* __restrict__ fcos,
                      const float* __restrict__ fsin, const int* __restrict__ cur_start,
                      bf16* __restrict__ Qb, bf16* __restrict__ Kc, bf16* __restrict__ VcT) {
  const int t   = blockIdx.x;
  const int tid = threadIdx.x;
  const float* rq = qkv + (size_t)t * 3 * DIM_;
  const float* rk = rq + DIM_;
  const float* rv = rq + 2 * DIM_;

  __shared__ float red[2][256];
  float aq = 0.f, ak = 0.f;
  for (int i = tid; i < DIM_; i += 256) {
    float a = rq[i]; aq += a * a;
    float b = rk[i]; ak += b * b;
  }
  red[0][tid] = aq; red[1][tid] = ak;
  __syncthreads();
  for (int s = 128; s > 0; s >>= 1) {
    if (tid < s) { red[0][tid] += red[0][tid + s]; red[1][tid] += red[1][tid + s]; }
    __syncthreads();
  }
  const float invq = rsqrtf(red[0][0] / DIM_ + 1e-6f);
  const float invk = rsqrtf(red[1][0] / DIM_ + 1e-6f);

  const int start_frame = cur_start[0] / FRAME_;
  const int ff = t / FRAME_;
  const int hh = (t / 52) % 30;
  const int ww = t % 52;

  for (int p = tid; p < NH_ * 64; p += 256) {
    int h = p >> 6, i = p & 63;
    int frow = (i < 22) ? (start_frame + ff) : ((i < 43) ? hh : ww);
    float c = fcos[frow * 64 + i];
    float s = fsin[frow * 64 + i];
    int base = h * HD_ + 2 * i;
    float qr = rq[base]     * invq * nqw[base];
    float qi = rq[base + 1] * invq * nqw[base + 1];
    float kr = rk[base]     * invk * nkw[base];
    float ki = rk[base + 1] * invk * nkw[base + 1];
    bf16* qd = Qb + (size_t)h * SP * HD_ + (size_t)t * HD_ + 2 * i;
    qd[0] = (bf16)(qr * c - qi * s);
    qd[1] = (bf16)(qr * s + qi * c);
    bf16* kd = Kc + (size_t)h * LP * HD_ + (size_t)(NEWOFF + t) * HD_ + 2 * i;
    kd[0] = (bf16)(kr * c - ki * s);
    kd[1] = (bf16)(kr * s + ki * c);
  }
  for (int i = tid; i < DIM_; i += 256) {
    int h = i / HD_, d = i % HD_;
    VcT[(size_t)h * HD_ * LP + (size_t)d * LP + NEWOFF + t] = (bf16)rv[i];
  }
}

// ---------------------------------------------------------------------------
// Flash attention. Block = 4 waves, one 16-row Q tile each, all sharing one
// head. K/V 32-key tiles are staged block-wide into LDS (double-buffered,
// async global->LDS), so each tile is fetched from global exactly once and
// feeds 64 q-rows: ~64 FLOP/byte of staged data. Online softmax uses 16-lane
// xor-shuffle reductions matching the C-fragment row layout; P is
// re-fragmented C->A through a per-wave LDS buffer guarded by s_wait_dscnt.
// ---------------------------------------------------------------------------
__global__ __launch_bounds__(128)
void attention_kernel(const bf16* __restrict__ Qb, const bf16* __restrict__ Kc,
                      const bf16* __restrict__ VcT, bf16* __restrict__ Ob) {
  const int h    = blockIdx.x;
  const int tid  = threadIdx.x;
  const int lane = tid & 31;
  const int wv   = tid >> 5;
  const int wt   = blockIdx.y * 4 + wv;
  const bool active = (wt < SP / 16);
  const int m0   = (active ? wt : (SP / 16 - 1)) * 16;   // clamp: EXEC stays all-1s
  const int lr   = lane & 15, lh = lane >> 4;

  __shared__ bf16 Kt[2][32 * HD_];   // [key in tile][dim]
  __shared__ bf16 Vt[2][HD_ * 32];   // [dim][key in tile]
  __shared__ bf16 Pb[4][16 * 32];
  bf16* pbuf = Pb[wv];

  const bf16* Kh = Kc + (size_t)h * LP * HD_;
  const bf16* Vh = VcT + (size_t)h * HD_ * LP;

  // Stage one 32-key tile (8KB K + 8KB V) block-wide: 8 async b128 per thread.
  auto stage = [&](int jb, int buf) {
    const bf16* kg = Kh + (size_t)jb * 32 * HD_;   // contiguous 8KB slice
#pragma unroll
    for (int it = 0; it < 4; it++) {
      int c = tid + it * 128;                      // 512 chunks of 8 elems
      async_copy16(Kt[buf] + c * 8, kg + c * 8);
    }
#pragma unroll
    for (int it = 0; it < 4; it++) {
      int c = tid + it * 128;
      int d = c >> 2, part = c & 3;
      async_copy16(Vt[buf] + d * 32 + part * 8,
                   Vh + (size_t)d * LP + (size_t)jb * 32 + part * 8);
    }
  };

  // Q fragments (loaded once from global).
  Frag qa[4];
  const bf16* qrow = Qb + ((size_t)h * SP + m0 + lr) * HD_ + lh * 8;
#pragma unroll
  for (int ds = 0; ds < 4; ds++) {
    qa[ds].h[0] = *(const v8bf*)(qrow + ds * 32);
    qa[ds].h[1] = *(const v8bf*)(qrow + ds * 32 + 16);
  }

  v8f o[8];
#pragma unroll
  for (int t = 0; t < 8; t++) o[t] = {};
  float mrow[8], lrow[8];
#pragma unroll
  for (int r = 0; r < 8; r++) { mrow[r] = -3.0e38f; lrow[r] = 0.0f; }

  const float scale = 0.08838834764831845f;  // 1/sqrt(128)
  const int NB = LP / 32;

  stage(0, 0);
  wait_async();
  __syncthreads();

  for (int jb = 0; jb < NB; jb++) {
    const int buf = jb & 1;
    if (jb + 1 < NB) stage(jb + 1, buf ^ 1);   // overlaps with compute below

    // ---- S = Q K^T : 16x32 logits (two 16x16 fragments), K frags from LDS --
    v8f s[2]; s[0] = {}; s[1] = {};
#pragma unroll
    for (int g = 0; g < 2; g++) {
      const bf16* kr = Kt[buf] + (g * 16 + lr) * HD_ + lh * 8;
      Frag kb[4];
#pragma unroll
      for (int ds = 0; ds < 4; ds++) {
        kb[ds].h[0] = *(const v8bf*)(kr + ds * 32);
        kb[ds].h[1] = *(const v8bf*)(kr + ds * 32 + 16);
      }
#pragma unroll
      for (int ds = 0; ds < 4; ds++)
        s[g] = __builtin_amdgcn_wmma_f32_16x16x32_bf16(
            false, qa[ds].v, false, kb[ds].v, (short)0, s[g], false, false);
    }
    // ---- scale + pad mask ----
#pragma unroll
    for (int g = 0; g < 2; g++) {
      int col = jb * 32 + g * 16 + lr;
      float mask = (col < LKV) ? 0.0f : -3.0e38f;
#pragma unroll
      for (int r = 0; r < 8; r++) s[g][r] = s[g][r] * scale + mask;
    }
    // ---- online softmax ----
#pragma unroll
    for (int r = 0; r < 8; r++) {
      float rm = fmaxf(s[0][r], s[1][r]);
#pragma unroll
      for (int off = 8; off >= 1; off >>= 1) rm = fmaxf(rm, __shfl_xor(rm, off, 32));
      float mnew  = fmaxf(mrow[r], rm);
      float alpha = __expf(mrow[r] - mnew);
      float p0 = __expf(s[0][r] - mnew);
      float p1 = __expf(s[1][r] - mnew);
      s[0][r] = p0; s[1][r] = p1;
      float rs = p0 + p1;
#pragma unroll
      for (int off = 8; off >= 1; off >>= 1) rs += __shfl_xor(rs, off, 32);
      lrow[r] = lrow[r] * alpha + rs;
      mrow[r] = mnew;
#pragma unroll
      for (int t = 0; t < 8; t++) o[t][r] *= alpha;
    }
    // ---- P: C-layout -> LDS -> A-layout ----
#pragma unroll
    for (int g = 0; g < 2; g++)
#pragma unroll
      for (int r = 0; r < 8; r++)
        pbuf[(r + 8 * lh) * 32 + g * 16 + lr] = (bf16)s[g][r];
    asm volatile("s_wait_dscnt 0" ::: "memory");
    Frag pa;
    pa.h[0] = *(const v8bf*)(pbuf + lr * 32 + lh * 8);
    pa.h[1] = *(const v8bf*)(pbuf + lr * 32 + 16 + lh * 8);
    // ---- O += P V (V frags from LDS, contiguous) ----
#pragma unroll
    for (int dn = 0; dn < 8; dn++) {
      const bf16* vr = Vt[buf] + (dn * 16 + lr) * 32 + lh * 8;
      Frag vb;
      vb.h[0] = *(const v8bf*)(vr);
      vb.h[1] = *(const v8bf*)(vr + 16);
      o[dn] = __builtin_amdgcn_wmma_f32_16x16x32_bf16(
          false, pa.v, false, vb.v, (short)0, o[dn], false, false);
    }

    wait_async();      // next tile's copies have landed behind the compute
    __syncthreads();
  }

  if (active) {
#pragma unroll
    for (int dn = 0; dn < 8; dn++) {
#pragma unroll
      for (int r = 0; r < 8; r++) {
        int row = m0 + r + 8 * lh;
        float val = o[dn][r] / lrow[r];
        Ob[(size_t)row * DIM_ + h * HD_ + dn * 16 + lr] = (bf16)val;
      }
    }
  }
}

// ---------------------------------------------------------------------------
// host side
// ---------------------------------------------------------------------------
extern "C" void kernel_launch(void* const* d_in, const int* in_sizes, int n_in,
                              void* d_out, int out_size, void* d_ws, size_t ws_size,
                              hipStream_t stream) {
  (void)in_sizes; (void)n_in; (void)out_size; (void)ws_size;
  const float* x    = (const float*)d_in[0];
  const float* Wq   = (const float*)d_in[1];
  const float* bq   = (const float*)d_in[2];
  const float* Wk   = (const float*)d_in[3];
  const float* bk   = (const float*)d_in[4];
  const float* Wv   = (const float*)d_in[5];
  const float* bv   = (const float*)d_in[6];
  const float* Wo   = (const float*)d_in[7];
  const float* bo   = (const float*)d_in[8];
  const float* nqw  = (const float*)d_in[9];
  const float* nkw  = (const float*)d_in[10];
  const float* ck   = (const float*)d_in[11];
  const float* cv   = (const float*)d_in[12];
  const float* fcos = (const float*)d_in[13];
  const float* fsin = (const float*)d_in[14];
  const int* cur_start = (const int*)d_in[17];
  float* out = (float*)d_out;

  char* wsp = (char*)d_ws;
  size_t off = 0;
  auto take = [&](size_t bytes) -> char* {
    off = (off + 255) & ~(size_t)255;
    char* p = wsp + off;
    off += bytes;
    return p;
  };
  bf16*  xb      = (bf16*)take((size_t)SP * DIM_ * 2);
  bf16*  wqkv    = (bf16*)take((size_t)3 * DIM_ * DIM_ * 2);
  bf16*  wob     = (bf16*)take((size_t)DIM_ * DIM_ * 2);
  float* biasqkv = (float*)take((size_t)3 * DIM_ * 4);
  float* qkv_raw = (float*)take((size_t)SP * 3 * DIM_ * 4);
  bf16*  Qb      = (bf16*)take((size_t)NH_ * SP * HD_ * 2);
  bf16*  Kc      = (bf16*)take((size_t)NH_ * LP * HD_ * 2);
  bf16*  VcT     = (bf16*)take((size_t)NH_ * HD_ * LP * 2);
  bf16*  Ob      = (bf16*)take((size_t)SP * DIM_ * 2);

  const int WN = DIM_ * DIM_;
  cvt_f32_to_bf16<<<(WN + 255) / 256, 256, 0, stream>>>(Wq, wqkv,           WN);
  cvt_f32_to_bf16<<<(WN + 255) / 256, 256, 0, stream>>>(Wk, wqkv + WN,      WN);
  cvt_f32_to_bf16<<<(WN + 255) / 256, 256, 0, stream>>>(Wv, wqkv + 2 * WN,  WN);
  cvt_f32_to_bf16<<<(WN + 255) / 256, 256, 0, stream>>>(Wo, wob,            WN);
  cvt_x_pad<<<(SP * DIM_ + 255) / 256, 256, 0, stream>>>(x, xb);
  concat_bias<<<(3 * DIM_ + 255) / 256, 256, 0, stream>>>(bq, bk, bv, biasqkv);

  // QKV projection: [SP x 1536] @ [4608 x 1536]^T
  gemm_bf16_wmma<<<dim3(3 * DIM_ / 256, SP / 16), 128, 0, stream>>>(
      xb, wqkv, biasqkv, qkv_raw, DIM_, 3 * DIM_, SP);

  norm_rope_kernel<<<S_TOK, 256, 0, stream>>>(qkv_raw, nqw, nkw, fcos, fsin,
                                              cur_start, Qb, Kc, VcT);

  gather_cache<<<(NH_ * NEWOFF * HD_ + 255) / 256, 256, 0, stream>>>(ck, cv, Kc, VcT);
  pad_kernel<<<(NH_ * 16 * HD_ + 255) / 256, 256, 0, stream>>>(Kc, VcT, Qb);

  attention_kernel<<<dim3(NH_, (SP / 16 + 3) / 4), 128, 0, stream>>>(Qb, Kc, VcT, Ob);

  // Output projection: [SP x 1536] @ [1536 x 1536]^T, store first 1560 rows
  gemm_bf16_wmma<<<dim3(DIM_ / 256, SP / 16), 128, 0, stream>>>(
      Ob, wob, bo, out, DIM_, DIM_, S_TOK);
}
